// RNN_28991029248516
// MI455X (gfx1250) — matrix operand order repaired and checked by reading, not compiled, not run on hardware
//
#include <hip/hip_runtime.h>
#include <math.h>

// ---------------------------------------------------------------------------
// 2-layer tanh RNN for MI455X (gfx1250, wave32, WMMA).
//   Phase A: z = x @ W_ih^T + (b_ih + b_hh)   -> big bf16 WMMA GEMM, parallel
//   Phase B: h_t = tanh(z_t + h_{t-1} @ W_hh^T) -> persistent 1-WGP kernel,
//            W_hh pinned in VGPRs as WMMA B-fragments, h double-buffered in LDS
// Workspace usage: ~131 MB (x_bf16 32MB | y0_bf16 32MB | z 64MB | weights 2MB).
// ---------------------------------------------------------------------------

#define BB 32
#define TT 1024
#define DD 512
#define HH 512
#define RR (BB * TT)

typedef __bf16 bf16_t;
typedef __attribute__((ext_vector_type(16))) __bf16        v16bf;
typedef __attribute__((ext_vector_type(8)))  float         v8f;
typedef __attribute__((ext_vector_type(4)))  unsigned int  uvec4;

union Frag {
    v16bf v;
    uvec4 q[2];
};

__device__ __forceinline__ v8f wmma_bf16(v16bf a, v16bf b, v8f c) {
    // D = A(16x32 bf16) * B(32x16 bf16) + C(16x16 f32)
    return __builtin_amdgcn_wmma_f32_16x16x32_bf16(
        /*neg_a=*/false, a, /*neg_b=*/false, b,
        /*c_mod=*/(short)0, c, /*reuse_a=*/false, /*reuse_b=*/false);
}

// ---------------------------------------------------------------------------
// fp32 -> bf16 streaming convert (grid-stride)
// ---------------------------------------------------------------------------
__global__ void cvt_f32_to_bf16(const float* __restrict__ s,
                                bf16_t* __restrict__ d, int n) {
    for (int i = blockIdx.x * blockDim.x + threadIdx.x; i < n;
         i += gridDim.x * blockDim.x)
        d[i] = (bf16_t)s[i];
}

__global__ void add_bias(const float* __restrict__ a,
                         const float* __restrict__ b,
                         float* __restrict__ o, int n) {
    int i = blockIdx.x * blockDim.x + threadIdx.x;
    if (i < n) o[i] = a[i] + b[i];
}

// ---------------------------------------------------------------------------
// Phase A: Z[t, b, n] = sum_k A[b*T+t, k] * W[n, k] + bias[n]
// A: [RR, 512] bf16 row-major. W: [512, 512] bf16 row-major (output-major, so
// B-matrix (=W^T) fragments are contiguous row reads of W).
// Each wave computes a 32x32 output tile over the full K=512 (16 k-steps,
// 64 WMMAs). 8 waves / block; consecutive waves share a tileM row for L0 reuse.
// ---------------------------------------------------------------------------
__global__ __launch_bounds__(256) void gemm_ih_bf16(
    const bf16_t* __restrict__ A, const bf16_t* __restrict__ W,
    const float* __restrict__ bias, float* __restrict__ Z) {
    const int lane  = threadIdx.x & 31;
    const int wv    = threadIdx.x >> 5;
    const int wid   = blockIdx.x * 8 + wv;
    const int tileN = wid & 15;   // 16 tiles of 32 cols
    const int tileM = wid >> 4;   // 1024 tiles of 32 rows
    const int l15   = lane & 15;
    const int khalf = lane >> 4;

    v8f zero = {};
    v8f acc[2][2];
    acc[0][0] = zero; acc[0][1] = zero; acc[1][0] = zero; acc[1][1] = zero;

#pragma unroll 4
    for (int kk = 0; kk < 16; ++kk) {
        const int kA = kk * 32 + khalf * 8;   // A chunks: k..k+7 and k+16..k+23
        const int kB = kk * 32 + khalf * 8;   // B chunks: k..k+7 and k+16..k+23
        Frag a[2], b[2];
#pragma unroll
        for (int mt = 0; mt < 2; ++mt) {
            const bf16_t* pa =
                A + (size_t)(tileM * 32 + mt * 16 + l15) * 512 + kA;
            a[mt].q[0] = *(const uvec4*)pa;
            a[mt].q[1] = *(const uvec4*)(pa + 16);
        }
#pragma unroll
        for (int nt = 0; nt < 2; ++nt) {
            const bf16_t* pw =
                W + (size_t)(tileN * 32 + nt * 16 + l15) * 512 + kB;
            b[nt].q[0] = *(const uvec4*)pw;
            b[nt].q[1] = *(const uvec4*)(pw + 16);
        }
#pragma unroll
        for (int mt = 0; mt < 2; ++mt)
#pragma unroll
            for (int nt = 0; nt < 2; ++nt)
                acc[mt][nt] = wmma_bf16(a[mt].v, b[nt].v, acc[mt][nt]);
    }

    // Epilogue: +bias, scatter to Z in [T, B, H] order for the recurrence.
#pragma unroll
    for (int mt = 0; mt < 2; ++mt) {
#pragma unroll
        for (int nt = 0; nt < 2; ++nt) {
            const int n  = tileN * 32 + nt * 16 + l15;
            const float bn = bias[n];
#pragma unroll
            for (int r = 0; r < 8; ++r) {
                const int m  = tileM * 32 + mt * 16 + r + 8 * khalf;
                const int bt = m >> 10;    // batch
                const int tt = m & 1023;   // time
                Z[((size_t)tt * BB + bt) * HH + n] = acc[mt][nt][r] + bn;
            }
        }
    }
}

// ---------------------------------------------------------------------------
// Phase B: sequential recurrence, one persistent workgroup (32 waves on 1 WGP).
// Wave w owns output columns [16w, 16w+16). Its W_hh B-fragments (16 k-steps,
// 128 VGPRs) are loaded ONCE and stay in registers for all 1024 steps.
// h is double-buffered in LDS (2 x 32KB); one barrier per step.
// z_t is fetched into registers BEFORE the WMMA chain so its L2 latency hides
// under the 32 WMMAs; next step's z is prefetched (global_prefetch_b8).
// ---------------------------------------------------------------------------
__global__ __launch_bounds__(1024) void rnn_recurrence(
    const float* __restrict__ Z,     // [TT, BB, HH]
    const bf16_t* __restrict__ Whh,  // [HH, HH] row-major
    bf16_t* __restrict__ Ybf,        // [BB, TT, HH] bf16 out (feeds next GEMM) or nullptr
    float* __restrict__ Yf,          // [BB, TT, HH] f32 out (final y1) or nullptr
    float* __restrict__ Hlast) {     // [BB, HH] final hidden
    __shared__ __align__(16) bf16_t hbuf[2][BB * HH];  // 64 KB of 320 KB LDS

    const int lane  = threadIdx.x & 31;
    const int wv    = threadIdx.x >> 5;
    const int l15   = lane & 15;
    const int khalf = lane >> 4;
    const int nCol  = wv * 16 + l15;

    // Pin this wave's W_hh slab in VGPRs as ready-to-issue B fragments.
    Frag bw[16];
#pragma unroll
    for (int kk = 0; kk < 16; ++kk) {
        const bf16_t* pw = Whh + (size_t)nCol * HH + kk * 32 + khalf * 8;
        bw[kk].q[0] = *(const uvec4*)pw;
        bw[kk].q[1] = *(const uvec4*)(pw + 16);
    }

    // h_0 = 0
    for (int i = threadIdx.x; i < BB * HH; i += 1024) hbuf[0][i] = (bf16_t)0.0f;
    __syncthreads();

    for (int t = 0; t < TT; ++t) {
        const bf16_t* hc = hbuf[t & 1];
        bf16_t* hn       = hbuf[(t & 1) ^ 1];
        const float* zp  = Z + (size_t)t * (BB * HH);

        // Issue z_t loads first: independent of h, they complete under WMMAs.
        float zr0[8], zr1[8];
#pragma unroll
        for (int r = 0; r < 8; ++r) {
            const int m0 = r + 8 * khalf;
            zr0[r] = zp[m0 * HH + nCol];
            zr1[r] = zp[(m0 + 16) * HH + nCol];
        }
        // Hint next step's z toward the caches while we compute.
        if (t + 1 < TT) {
            __builtin_prefetch(zp + BB * HH + (size_t)khalf * 8 * HH + nCol, 0, 0);
            __builtin_prefetch(zp + BB * HH + (size_t)(16 + khalf * 8) * HH + nCol, 0, 0);
        }

        v8f c0 = {};
        v8f c1 = {};
#pragma unroll
        for (int kk = 0; kk < 16; ++kk) {
            Frag a0, a1;
            const bf16_t* p0 = hc + (size_t)l15 * HH + kk * 32 + khalf * 8;
            a0.q[0] = *(const uvec4*)p0;
            a0.q[1] = *(const uvec4*)(p0 + 16);
            const bf16_t* p1 = p0 + 16 * HH;   // rows 16..31
            a1.q[0] = *(const uvec4*)p1;
            a1.q[1] = *(const uvec4*)(p1 + 16);
            c0 = wmma_bf16(a0.v, bw[kk].v, c0);
            c1 = wmma_bf16(a1.v, bw[kk].v, c1);
        }

        // Fused epilogue: + z_t, tanh, write next-h (LDS) + outputs (global).
#pragma unroll
        for (int r = 0; r < 8; ++r) {
            const int m0 = r + 8 * khalf;  // C/D layout: lanes 16-31 -> M=8+r
            const int m1 = m0 + 16;
            float v0 = tanhf(c0[r] + zr0[r]);
            float v1 = tanhf(c1[r] + zr1[r]);
            hn[m0 * HH + nCol] = (bf16_t)v0;
            hn[m1 * HH + nCol] = (bf16_t)v1;
            if (Ybf) {
                Ybf[((size_t)m0 * TT + t) * HH + nCol] = (bf16_t)v0;
                Ybf[((size_t)m1 * TT + t) * HH + nCol] = (bf16_t)v1;
            }
            if (Yf) {
                Yf[((size_t)m0 * TT + t) * HH + nCol] = v0;
                Yf[((size_t)m1 * TT + t) * HH + nCol] = v1;
            }
            if (t == TT - 1) {
                Hlast[m0 * HH + nCol] = v0;
                Hlast[m1 * HH + nCol] = v1;
            }
        }
        __syncthreads();
    }
}

// ---------------------------------------------------------------------------
extern "C" void kernel_launch(void* const* d_in, const int* in_sizes, int n_in,
                              void* d_out, int out_size, void* d_ws,
                              size_t ws_size, hipStream_t stream) {
    (void)in_sizes; (void)n_in; (void)out_size; (void)ws_size;

    const float* x     = (const float*)d_in[0];
    const float* w_ih0 = (const float*)d_in[1];
    const float* w_hh0 = (const float*)d_in[2];
    const float* b_ih0 = (const float*)d_in[3];
    const float* b_hh0 = (const float*)d_in[4];
    const float* w_ih1 = (const float*)d_in[5];
    const float* w_hh1 = (const float*)d_in[6];
    const float* b_ih1 = (const float*)d_in[7];
    const float* b_hh1 = (const float*)d_in[8];
    float* out = (float*)d_out;

    // Workspace carve (all 16B-aligned offsets).
    char*   ws  = (char*)d_ws;
    bf16_t* xb  = (bf16_t*)ws;                        // 32 MB: x in bf16
    bf16_t* yb  = (bf16_t*)(ws + (33554432ull));      // 32 MB: y0 in bf16
    float*  z   = (float*)(ws + (67108864ull));       // 64 MB: z, reused per layer
    bf16_t* wi0 = (bf16_t*)(ws + (134217728ull));     // 4 x 512 KB weights
    bf16_t* wh0 = wi0 + 262144;
    bf16_t* wi1 = wh0 + 262144;
    bf16_t* wh1 = wi1 + 262144;
    float*  bs0 = (float*)(ws + 134217728ull + 4ull * 524288ull);
    float*  bs1 = bs0 + 512;

    // Precision prep (one streaming pass).
    cvt_f32_to_bf16<<<4096, 256, 0, stream>>>(x, xb, RR * DD);
    cvt_f32_to_bf16<<<512, 256, 0, stream>>>(w_ih0, wi0, HH * DD);
    cvt_f32_to_bf16<<<512, 256, 0, stream>>>(w_hh0, wh0, HH * HH);
    cvt_f32_to_bf16<<<512, 256, 0, stream>>>(w_ih1, wi1, HH * HH);
    cvt_f32_to_bf16<<<512, 256, 0, stream>>>(w_hh1, wh1, HH * HH);
    add_bias<<<2, 256, 0, stream>>>(b_ih0, b_hh0, bs0, HH);
    add_bias<<<2, 256, 0, stream>>>(b_ih1, b_hh1, bs1, HH);

    // Layer 0: bulk ih-projection, then sequential recurrence.
    gemm_ih_bf16<<<2048, 256, 0, stream>>>(xb, wi0, bs0, z);
    rnn_recurrence<<<1, 1024, 0, stream>>>(z, wh0, yb, (float*)nullptr,
                                           out + 16777216);  // h0

    // Layer 1.
    gemm_ih_bf16<<<2048, 256, 0, stream>>>(yb, wi1, bs1, z);
    rnn_recurrence<<<1, 1024, 0, stream>>>(z, wh1, (bf16_t*)nullptr, out,
                                           out + 16777216 + BB * HH);  // h1
}
